// DiffusionActionHead_65859028517381
// MI455X (gfx1250) — matrix operand order repaired and checked by reading, not compiled, not run on hardware
//
#include <hip/hip_runtime.h>
#include <math.h>

#define HDIM 256
#define ACT_D 4
#define BATCH 32768
#define ROWS_PER_WG 128
#define NWG (BATCH / ROWS_PER_WG)
#define THREADS 256
#define HPAD 264  // padded LDS row stride (halves) to spread banks

#define FRAG_HALVES 512                 // 32 lanes * 16 halves per B fragment
#define MAT_HALVES (128 * FRAG_HALVES)  // 65536 halves per 256x256 matrix
#define WHALF_HALVES (MAT_HALVES / 2)   // 8 N-tiles staged at a time (64KB)

// LDS layout (halves): hP | hQ | Wbuf0 | Wbuf1 | xs(float)
#define LDS_HP 0
#define LDS_HQ (ROWS_PER_WG * HPAD)
#define LDS_W0 (2 * ROWS_PER_WG * HPAD)
#define LDS_W1 (2 * ROWS_PER_WG * HPAD + WHALF_HALVES)
#define LDS_HALVES (2 * ROWS_PER_WG * HPAD + 2 * WHALF_HALVES)
#define DYN_LDS_BYTES (LDS_HALVES * 2 + ROWS_PER_WG * ACT_D * 4)  // 268,288 B

typedef __attribute__((ext_vector_type(16))) _Float16 v16h;
typedef __attribute__((ext_vector_type(8))) _Float16 v8h;
typedef __attribute__((ext_vector_type(8))) float v8f;
typedef __attribute__((ext_vector_type(4))) int v4i;

#define GLOBAL_AS __attribute__((address_space(1)))
#define LDS_AS __attribute__((address_space(3)))

enum { M_BIAS_F16 = 0, M_STORE_F32, M_GELU, M_GELU_SKIP, M_GELU_RES };

__device__ __forceinline__ float gelu_f(float x) {
  return 0.5f * x * (1.0f + erff(x * 0.70710678118654752f));
}

__device__ __forceinline__ unsigned mix32(unsigned x) {
  x ^= x >> 16; x *= 0x7feb352du; x ^= x >> 15; x *= 0x846ca68bu; x ^= x >> 16;
  return x;
}
__device__ __forceinline__ float gauss1(unsigned salt, unsigned idx) {
  unsigned h1 = mix32(idx * 0x9E3779B9u ^ salt * 0x85EBCA6Bu ^ 0xC2B2AE35u);
  unsigned h2 = mix32(h1 + 0x632BE59Bu);
  float u1 = (float)((h1 >> 8) + 1u) * (1.0f / 16777216.0f);
  float u2 = (float)(h2 >> 8) * (1.0f / 16777216.0f);
  return sqrtf(-2.0f * logf(u1)) * cosf(6.283185307f * u2);
}

// ---------------------------------------------------------------------------
// CDNA5 async global->LDS copy of one 64KB weight half (16B per thread x16).
// Tracked by ASYNCcnt; overlapped with WMMA compute. Falls back to a
// synchronous VGPR round-trip if the toolchain lacks the builtin.
// ---------------------------------------------------------------------------
__device__ __forceinline__ void async_wait0() {
#if __has_builtin(__builtin_amdgcn_s_wait_asynccnt)
  __builtin_amdgcn_s_wait_asynccnt(0);
#else
  asm volatile("s_wait_asynccnt 0x0" ::: "memory");
#endif
}

__device__ __forceinline__ void async_copy_64k(const _Float16* __restrict__ g,
                                               _Float16* l, int tid) {
#if __has_builtin(__builtin_amdgcn_global_load_async_to_lds_b128)
  GLOBAL_AS v4i* gp = (GLOBAL_AS v4i*)g;
  LDS_AS v4i* lp = (LDS_AS v4i*)l;
#pragma unroll
  for (int i = 0; i < 16; ++i) {
    __builtin_amdgcn_global_load_async_to_lds_b128(gp + (tid + i * 256),
                                                   lp + (tid + i * 256), 0, 0);
  }
#else
  const uint4* s = (const uint4*)g;
  uint4* d = (uint4*)l;
#pragma unroll
  for (int i = 0; i < 16; ++i) d[tid + i * 256] = s[tid + i * 256];
#endif
}

// ---------------------------------------------------------------------------
// 8 N-tiles (one staged weight half) of the 128x256x256 layer GEMM.
// ---------------------------------------------------------------------------
template <int MODE>
__device__ __forceinline__ void compute_half(
    int halfIdx, const _Float16* Wl, const v16h* Af, _Float16* Obuf,
    const float* __restrict__ bias, const float* __restrict__ PSg,
    const float* __restrict__ tvS, const float* __restrict__ wsx,
    const float* xs, float* __restrict__ gout, int wgBase, int tid) {
  const int lane = tid & 31;
  const int wave = tid >> 5;
  for (int n = 0; n < 8; ++n) {
    const int nt = halfIdx * 8 + n;
    const int col = nt * 16 + (lane & 15);
    float bv = 0.0f;
    if constexpr (MODE != M_STORE_F32) bv = bias[col];
    v8f c;
#pragma unroll
    for (int i = 0; i < 8; ++i) c[i] = bv;
#pragma unroll
    for (int kt = 0; kt < 8; ++kt) {
      v16h b = *(const v16h*)(Wl + (n * 8 + kt) * FRAG_HALVES + lane * 16);
      c = __builtin_amdgcn_wmma_f32_16x16x32_f16(false, Af[kt], false, b,
                                                 (short)0, c, false, false);
    }
    // C/D layout: VGPR i holds row (i | (lane>=16 ? 8 : 0)), col = lane&15
#pragma unroll
    for (int i = 0; i < 8; ++i) {
      const int row = wave * 16 + i + ((lane & 16) ? 8 : 0);
      float v = c[i];
      if constexpr (MODE == M_BIAS_F16) {
        Obuf[row * HPAD + col] = (_Float16)v;
      } else if constexpr (MODE == M_STORE_F32) {
        gout[(size_t)(wgBase + row) * HDIM + col] = v;
      } else if constexpr (MODE == M_GELU) {
        Obuf[row * HPAD + col] = (_Float16)gelu_f(v);
      } else if constexpr (MODE == M_GELU_SKIP) {
        float skip = PSg[(size_t)(wgBase + row) * HDIM + col] + tvS[col] +
                     xs[row * ACT_D + 0] * wsx[0 * HDIM + col] +
                     xs[row * ACT_D + 1] * wsx[1 * HDIM + col] +
                     xs[row * ACT_D + 2] * wsx[2 * HDIM + col] +
                     xs[row * ACT_D + 3] * wsx[3 * HDIM + col];
        Obuf[row * HPAD + col] = (_Float16)(gelu_f(v) + skip);
      } else {  // M_GELU_RES
        float r = (float)Obuf[row * HPAD + col];
        Obuf[row * HPAD + col] = (_Float16)(gelu_f(v) + r);
      }
    }
  }
}

// ---------------------------------------------------------------------------
// One GEMM layer with double-buffered async weight pipeline.
// Precondition: Wb0 holds this layer's half0 (loaded by caller/previous layer)
// Postcondition: Wb0 holds Wnext's half0 (if Wnext != nullptr)
// ---------------------------------------------------------------------------
template <int MODE>
__device__ __forceinline__ void gemm_layer(
    const _Float16* __restrict__ Wpack, const _Float16* __restrict__ Wnext,
    const _Float16* Abuf, _Float16* Obuf, _Float16* Wb0, _Float16* Wb1,
    const float* __restrict__ bias, const float* __restrict__ PSg,
    const float* __restrict__ tvS, const float* __restrict__ wsx,
    const float* xs, float* __restrict__ gout, int wgBase, int tid) {
  const int lane = tid & 31;
  const int wave = tid >> 5;

  // A fragments: 16x32 f16 per K-tile. lane L<16 -> row L, K {0..7,16..23};
  // lane L>=16 -> row L-16, K {8..15,24..31}  (ISA 7.12.2 16-bit A layout)
  v16h Af[8];
  {
    const _Float16* ar =
        Abuf + (wave * 16 + (lane & 15)) * HPAD + ((lane & 16) ? 8 : 0);
#pragma unroll
    for (int kt = 0; kt < 8; ++kt) {
      v8h lo = *(const v8h*)(ar + kt * 32);
      v8h hi = *(const v8h*)(ar + kt * 32 + 16);
      Af[kt] = __builtin_shufflevector(lo, hi, 0, 1, 2, 3, 4, 5, 6, 7, 8, 9, 10,
                                       11, 12, 13, 14, 15);
    }
  }

  // prefetch half1 into Wb1 while computing half0 from Wb0
  async_copy_64k(Wpack + WHALF_HALVES, Wb1, tid);
  compute_half<MODE>(0, Wb0, Af, Obuf, bias, PSg, tvS, wsx, xs, gout, wgBase,
                     tid);
  async_wait0();
  __syncthreads();

  // prefetch next layer's half0 into Wb0 while computing half1 from Wb1
  if (Wnext) async_copy_64k(Wnext, Wb0, tid);
  compute_half<MODE>(1, Wb1, Af, Obuf, bias, PSg, tvS, wsx, xs, gout, wgBase,
                     tid);
  async_wait0();
  __syncthreads();
}

// ---------------------------------------------------------------------------
// Repack 10 f32 256x256 weight matrices into WMMA B-fragment f16 layout:
// fragment f = nt*8+kt; lane L<16: (K=kt*32+(i<8?i:i+8), N=nt*16+L); L>=16: K += 8.
// ---------------------------------------------------------------------------
__global__ __launch_bounds__(256) void repack_kernel(
    const float* __restrict__ cond_w, const float* __restrict__ blk0_w1,
    const float* __restrict__ blk0_skip_w, const float* __restrict__ blk0_w2,
    const float* __restrict__ blks_w1, const float* __restrict__ blks_w2,
    _Float16* __restrict__ pack) {
  int gid = blockIdx.x * blockDim.x + threadIdx.x;
  if (gid >= 10 * 128 * 32) return;
  int m = gid / 4096;
  int f = (gid / 32) & 127;
  int lane = gid & 31;
  int nt = f >> 3, kt = f & 7;
  const float* src;
  if (m == 0) src = cond_w;
  else if (m == 1) src = blk0_w1;
  else if (m == 2) src = blk0_skip_w;
  else if (m == 3) src = blk0_w2;
  else if (m < 7) src = blks_w1 + (size_t)(m - 4) * HDIM * HDIM;
  else src = blks_w2 + (size_t)(m - 7) * HDIM * HDIM;
  int N = nt * 16 + (lane & 15);
  int kbase = kt * 32 + ((lane & 16) ? 8 : 0);
  _Float16* dst = pack + (size_t)m * MAT_HALVES + f * FRAG_HALVES + lane * 16;
#pragma unroll
  for (int i = 0; i < 16; ++i)
    dst[i] = (_Float16)src[(size_t)(kbase + (i < 8 ? i : i + 8)) * HDIM + N];
}

// ---------------------------------------------------------------------------
// Per-step time-embedding vectors for all 10 steps (tiny; one WG).
// ---------------------------------------------------------------------------
__global__ __launch_bounds__(256) void temb_kernel(
    const float* __restrict__ time_w, const float* __restrict__ time_b,
    const float* __restrict__ blk0_w1, const float* __restrict__ blk0_b1,
    const float* __restrict__ blk0_skip_w,
    const float* __restrict__ blk0_skip_b, float* __restrict__ tv1,
    float* __restrict__ tvS) {
  __shared__ float e[HDIM];
  __shared__ float temb[HDIM];
  const int j = threadIdx.x;
  for (int s = 0; s < 10; ++s) {
    float t = (float)(90 - 10 * s);
    float fr = expf((float)(j & 127) * (-9.210340371976184f / 127.0f));
    e[j] = (j < 128) ? sinf(t * fr) : cosf(t * fr);
    __syncthreads();
    float acc = time_b[j];
    for (int k = 0; k < HDIM; ++k) acc += e[k] * time_w[k * HDIM + j];
    float tb = gelu_f(acc);
    __syncthreads();
    temb[j] = tb;
    __syncthreads();
    float a1 = blk0_b1[j], aS = blk0_skip_b[j];
    for (int k = 0; k < HDIM; ++k) {
      float tv = temb[k];
      a1 += tv * blk0_w1[(size_t)(256 + k) * HDIM + j];
      aS += tv * blk0_skip_w[(size_t)(256 + k) * HDIM + j];
    }
    tv1[s * HDIM + j] = a1;
    tvS[s * HDIM + j] = aS;
    __syncthreads();
  }
}

__global__ __launch_bounds__(256) void init_x_kernel(float* __restrict__ xb) {
  int i = blockIdx.x * blockDim.x + threadIdx.x;
  if (i < BATCH * ACT_D) xb[i] = gauss1(0x3579u, (unsigned)i);
}

// ---------------------------------------------------------------------------
// Step-invariant precompute: cond = features@cond_w + cond_b (f16),
// P1 = cond@W1c, PS = cond@Wsc  (f32 in workspace)
// ---------------------------------------------------------------------------
__global__ __launch_bounds__(256) void prep_gemm_kernel(
    const float* __restrict__ features, const _Float16* __restrict__ pack,
    const float* __restrict__ cond_b, float* __restrict__ P1,
    float* __restrict__ PS) {
  extern __shared__ _Float16 smem[];
  _Float16* hP = smem + LDS_HP;
  _Float16* hQ = smem + LDS_HQ;
  _Float16* Wb0 = smem + LDS_W0;
  _Float16* Wb1 = smem + LDS_W1;
  const int tid = threadIdx.x;
  const int wgBase = blockIdx.x * ROWS_PER_WG;

  async_copy_64k(pack, Wb0, tid);  // first layer half0
  for (int i = 0; i < ROWS_PER_WG; ++i)
    hQ[i * HPAD + tid] = (_Float16)features[(size_t)(wgBase + i) * HDIM + tid];
  async_wait0();
  __syncthreads();

  gemm_layer<M_BIAS_F16>(pack + 0 * MAT_HALVES, pack + 1 * MAT_HALVES, hQ, hP,
                         Wb0, Wb1, cond_b, nullptr, nullptr, nullptr, nullptr,
                         nullptr, wgBase, tid);
  gemm_layer<M_STORE_F32>(pack + 1 * MAT_HALVES, pack + 2 * MAT_HALVES, hP,
                          nullptr, Wb0, Wb1, nullptr, nullptr, nullptr, nullptr,
                          nullptr, P1, wgBase, tid);
  gemm_layer<M_STORE_F32>(pack + 2 * MAT_HALVES, nullptr, hP, nullptr, Wb0,
                          Wb1, nullptr, nullptr, nullptr, nullptr, nullptr, PS,
                          wgBase, tid);
}

// ---------------------------------------------------------------------------
// One diffusion step, fully fused: 7 WMMA GEMM layers (async double-buffered
// weights) + final 256->4 projection + DDPM x update.
// ---------------------------------------------------------------------------
__global__ __launch_bounds__(256) void step_kernel(
    const _Float16* __restrict__ pack, const float* __restrict__ P1,
    const float* __restrict__ PS, const float* __restrict__ tv1,
    const float* __restrict__ tvS, const float* __restrict__ w1x,
    const float* __restrict__ wsx, const float* __restrict__ blk0_b2,
    const float* __restrict__ blks_b1, const float* __restrict__ blks_b2,
    const float* __restrict__ final_w, const float* __restrict__ final_b,
    float* __restrict__ xbuf, float* __restrict__ dout, float c_pred,
    float c_scale, float c_noise, int sidx, int last) {
  extern __shared__ _Float16 smem[];
  _Float16* hP = smem + LDS_HP;
  _Float16* hQ = smem + LDS_HQ;
  _Float16* Wb0 = smem + LDS_W0;
  _Float16* Wb1 = smem + LDS_W1;
  float* xs = (float*)(smem + LDS_HALVES);
  const int tid = threadIdx.x;
  const int lane = tid & 31, wave = tid >> 5;
  const int wgBase = blockIdx.x * ROWS_PER_WG;

  async_copy_64k(pack + 3 * MAT_HALVES, Wb0, tid);  // first layer half0

#pragma unroll
  for (int i = 0; i < 2; ++i) {
    int idx = tid + i * 256;  // 128 rows * 4
    xs[idx] = xbuf[(size_t)wgBase * ACT_D + idx];
  }
  __syncthreads();

  // a0 = gelu(P1 + tvec1 + x @ W1x) -> hQ   (wave-private rows, no barrier)
  for (int i = 0; i < 128; ++i) {
    int lin = i * 32 + lane;
    int row = wave * 16 + (lin >> 8);
    int col = lin & 255;
    float u = P1[(size_t)(wgBase + row) * HDIM + col] + tv1[col] +
              xs[row * ACT_D + 0] * w1x[0 * HDIM + col] +
              xs[row * ACT_D + 1] * w1x[1 * HDIM + col] +
              xs[row * ACT_D + 2] * w1x[2 * HDIM + col] +
              xs[row * ACT_D + 3] * w1x[3 * HDIM + col];
    hQ[row * HPAD + col] = (_Float16)gelu_f(u);
  }
  async_wait0();
  __syncthreads();

  // block0 second linear + full skip path
  gemm_layer<M_GELU_SKIP>(pack + 3 * MAT_HALVES, pack + 4 * MAT_HALVES, hQ, hP,
                          Wb0, Wb1, blk0_b2, PS, tvS, wsx, xs, nullptr, wgBase,
                          tid);
  // residual blocks 1..3
  for (int blk = 0; blk < 3; ++blk) {
    gemm_layer<M_GELU>(pack + (size_t)(4 + blk) * MAT_HALVES,
                       pack + (size_t)(7 + blk) * MAT_HALVES, hP, hQ, Wb0, Wb1,
                       blks_b1 + blk * HDIM, nullptr, nullptr, nullptr, nullptr,
                       nullptr, wgBase, tid);
    const _Float16* nw =
        (blk < 2) ? pack + (size_t)(5 + blk) * MAT_HALVES : nullptr;
    gemm_layer<M_GELU_RES>(pack + (size_t)(7 + blk) * MAT_HALVES, nw, hQ, hP,
                           Wb0, Wb1, blks_b2 + blk * HDIM, nullptr, nullptr,
                           nullptr, nullptr, nullptr, wgBase, tid);
  }

  // stage final_w / final_b into (now free) weight LDS
  float* fwl = (float*)Wb0;
  for (int i = tid; i < HDIM * ACT_D; i += 256) fwl[i] = final_w[i];
  if (tid < ACT_D) fwl[HDIM * ACT_D + tid] = final_b[tid];
  __syncthreads();

  // final projection (256 -> 4) + DDPM update, wave-private rows
#pragma unroll
  for (int o = 0; o < 2; ++o) {
    int oid = lane + o * 32;          // 0..63 outputs per wave
    int rl = wave * 16 + (oid >> 2);  // local row
    int j = oid & 3;
    float acc = fwl[HDIM * ACT_D + j];
    const _Float16* hrow = hP + rl * HPAD;
    for (int k8 = 0; k8 < HDIM / 8; ++k8) {
      v8h hv = *(const v8h*)(hrow + k8 * 8);
#pragma unroll
      for (int u = 0; u < 8; ++u)
        acc += (float)hv[u] * fwl[(k8 * 8 + u) * ACT_D + j];
    }
    int gRow = wgBase + rl;
    float xo = xs[rl * ACT_D + j];
    float xn = (xo - c_pred * acc) * c_scale;
    if (c_noise != 0.0f)
      xn += c_noise *
            gauss1(7000u + (unsigned)sidx, (unsigned)(gRow * ACT_D + j));
    xbuf[(size_t)gRow * ACT_D + j] = xn;
    if (last) dout[(size_t)gRow * ACT_D + j] = fminf(1.0f, fmaxf(-1.0f, xn));
  }
}

// ---------------------------------------------------------------------------
extern "C" void kernel_launch(void* const* d_in, const int* in_sizes, int n_in,
                              void* d_out, int out_size, void* d_ws,
                              size_t ws_size, hipStream_t stream) {
  (void)in_sizes; (void)n_in; (void)out_size; (void)ws_size;
  const float* features = (const float*)d_in[0];
  const float* cond_w = (const float*)d_in[1];
  const float* cond_b = (const float*)d_in[2];
  const float* time_w = (const float*)d_in[3];
  const float* time_b = (const float*)d_in[4];
  const float* blk0_w1 = (const float*)d_in[5];
  const float* blk0_b1 = (const float*)d_in[6];
  const float* blk0_w2 = (const float*)d_in[7];
  const float* blk0_b2 = (const float*)d_in[8];
  const float* blk0_skip_w = (const float*)d_in[9];
  const float* blk0_skip_b = (const float*)d_in[10];
  const float* blks_w1 = (const float*)d_in[11];
  const float* blks_b1 = (const float*)d_in[12];
  const float* blks_w2 = (const float*)d_in[13];
  const float* blks_b2 = (const float*)d_in[14];
  const float* final_w = (const float*)d_in[15];
  const float* final_b = (const float*)d_in[16];

  char* ws = (char*)d_ws;
  _Float16* pack = (_Float16*)ws;
  size_t off = (size_t)10 * MAT_HALVES * sizeof(_Float16);
  float* P1 = (float*)(ws + off);  off += (size_t)BATCH * HDIM * 4;
  float* PS = (float*)(ws + off);  off += (size_t)BATCH * HDIM * 4;
  float* tv1 = (float*)(ws + off); off += 10 * HDIM * 4;
  float* tvS = (float*)(ws + off); off += 10 * HDIM * 4;
  float* xbuf = (float*)(ws + off);

  (void)hipFuncSetAttribute((const void*)prep_gemm_kernel,
                            hipFuncAttributeMaxDynamicSharedMemorySize,
                            DYN_LDS_BYTES);
  (void)hipFuncSetAttribute((const void*)step_kernel,
                            hipFuncAttributeMaxDynamicSharedMemorySize,
                            DYN_LDS_BYTES);

  repack_kernel<<<160, 256, 0, stream>>>(cond_w, blk0_w1, blk0_skip_w, blk0_w2,
                                         blks_w1, blks_w2, pack);
  temb_kernel<<<1, 256, 0, stream>>>(time_w, time_b, blk0_w1, blk0_b1,
                                     blk0_skip_w, blk0_skip_b, tv1, tvS);
  init_x_kernel<<<(BATCH * ACT_D + 255) / 256, 256, 0, stream>>>(xbuf);
  prep_gemm_kernel<<<NWG, THREADS, DYN_LDS_BYTES, stream>>>(features, pack,
                                                            cond_b, P1, PS);

  float betas[100], alphas[100], acp[100];
  float c = 1.0f;
  for (int i = 0; i < 100; ++i) {
    betas[i] = 1e-4f + (0.02f - 1e-4f) * (float)i / 99.0f;
    alphas[i] = 1.0f - betas[i];
    c *= alphas[i];
    acp[i] = c;
  }
  for (int s = 0; s < 10; ++s) {
    int t = 90 - 10 * s;
    float cp = betas[t] / sqrtf(1.0f - acp[t]);
    float cs = 1.0f / sqrtf(alphas[t]);
    float cn = (t > 0) ? sqrtf(betas[t]) : 0.0f;
    step_kernel<<<NWG, THREADS, DYN_LDS_BYTES, stream>>>(
        pack, P1, PS, tv1 + s * HDIM, tvS + s * HDIM, blk0_w1 + 512 * HDIM,
        blk0_skip_w + 512 * HDIM, blk0_b2, blks_b1, blks_b2, final_w, final_b,
        xbuf, (float*)d_out, cp, cs, cn, s, (s == 9) ? 1 : 0);
  }
}
